// Perceiver_GWT_GWWM_13967233646904
// MI455X (gfx1250) — matrix operand order repaired and checked by reading, not compiled, not run on hardware
//
#include <hip/hip_runtime.h>
#include <hip/hip_bf16.h>

// ---------------------------------------------------------------------------
// Types for WMMA (gfx1250, wave32): V_WMMA_F32_16X16X32_BF16
// ---------------------------------------------------------------------------
typedef __bf16 v16bf __attribute__((ext_vector_type(16)));
typedef float  v8f   __attribute__((ext_vector_type(8)));

union Frag16 { v16bf v; uint4 q[2]; };

// A-matrix 16x32 bf16 fragment (ISA 7.12.2): lane l holds row m0+(l&15).
// lanes 0-15: elems 0..7 = K k0..k0+7, elems 8..15 = K k0+16..k0+23
// lanes 16-31: elems 0..7 = K k0+8..k0+15, elems 8..15 = K k0+24..k0+31
__device__ __forceinline__ v16bf load_a_frag(const __bf16* __restrict__ A, int lda,
                                             int row, int k0, int khalf) {
    Frag16 f;
    const __bf16* p = A + (long long)row * lda + k0 + khalf * 8;
    f.q[0] = *reinterpret_cast<const uint4*>(p);
    f.q[1] = *reinterpret_cast<const uint4*>(p + 16);
    return f.v;
}

// B-matrix 32x16 bf16 fragment: lane l holds column n0+(l&15) of B, i.e. row of
// W^T stored row-major [N][K]; lanes 0-15 hold K k0..k0+7,k0+16..k0+23, lanes 16-31 K k0+8..k0+15,k0+24..k0+31.
__device__ __forceinline__ v16bf load_b_frag(const __bf16* __restrict__ W, int ldw,
                                             int col, int k0, int khalf) {
    Frag16 f;
    const __bf16* p = W + (long long)col * ldw + k0 + khalf * 8;
    f.q[0] = *reinterpret_cast<const uint4*>(p);
    f.q[1] = *reinterpret_cast<const uint4*>(p + 16);
    return f.v;
}

__device__ __forceinline__ float gelu_exact(float x) {
    return 0.5f * x * (1.0f + erff(x * 0.70710678118654752f));
}

#define GOUT_F32        0
#define GOUT_BF16       1
#define GOUT_BF16_TRANS 2   // C stored transposed: out[n*ldc + m] (ldc == M)

// ---------------------------------------------------------------------------
// Fused bf16 GEMM: C[M,N] = act( scale * (A[M,K] @ W[N,K]^T) + bias + resid )
// One wave computes a 16x64 output tile (4 accumulators). grid.z batches heads.
// ---------------------------------------------------------------------------
__global__ void k_gemm(const __bf16* __restrict__ A, int lda, long long aBatch,
                       const __bf16* __restrict__ W, int ldw, long long wBatch,
                       const float* __restrict__ bias,
                       const float* __restrict__ resid, int ldr,
                       void* __restrict__ Cout, int ldc, long long cBatch,
                       int M, int N, int K, float scale, int mode, int doGelu) {
    const int z = blockIdx.z;
    A += (long long)z * aBatch;
    W += (long long)z * wBatch;

    const int lane  = threadIdx.x & 31;
    const int wid   = blockIdx.x * (blockDim.x >> 5) + (threadIdx.x >> 5);
    const int tN    = N >> 6;            // 64-wide wave tiles
    const int tM    = M >> 4;
    if (wid >= tM * tN) return;
    const int m0    = (wid / tN) << 4;
    const int n0    = (wid % tN) << 6;
    const int khalf = lane >> 4;
    const int l15   = lane & 15;
    const int arow  = m0 + l15;

    v8f acc[4] = {v8f{}, v8f{}, v8f{}, v8f{}};

    for (int k0 = 0; k0 < K; k0 += 32) {
        v16bf a  = load_a_frag(A, lda, arow, k0, khalf);
        v16bf b0 = load_b_frag(W, ldw, n0 +  0 + l15, k0, khalf);
        v16bf b1 = load_b_frag(W, ldw, n0 + 16 + l15, k0, khalf);
        v16bf b2 = load_b_frag(W, ldw, n0 + 32 + l15, k0, khalf);
        v16bf b3 = load_b_frag(W, ldw, n0 + 48 + l15, k0, khalf);
        acc[0] = __builtin_amdgcn_wmma_f32_16x16x32_bf16(false, a, false, b0, (short)0, acc[0], false, false);
        acc[1] = __builtin_amdgcn_wmma_f32_16x16x32_bf16(false, a, false, b1, (short)0, acc[1], false, false);
        acc[2] = __builtin_amdgcn_wmma_f32_16x16x32_bf16(false, a, false, b2, (short)0, acc[2], false, false);
        acc[3] = __builtin_amdgcn_wmma_f32_16x16x32_bf16(false, a, false, b3, (short)0, acc[3], false, false);
    }

    // C/D layout: VGPR r: lanes 0-15 -> (m0+r, n), lanes 16-31 -> (m0+8+r, n)
    const int mbase = m0 + khalf * 8;
    #pragma unroll
    for (int t = 0; t < 4; ++t) {
        const int n  = n0 + t * 16 + l15;
        const float bv = bias ? bias[n] : 0.0f;
        #pragma unroll
        for (int r = 0; r < 8; ++r) {
            const int m = mbase + r;
            float v = acc[t][r] * scale + bv;
            if (resid) v += resid[(long long)m * ldr + n];
            if (doGelu) v = gelu_exact(v);
            if (mode == GOUT_F32) {
                ((float*)Cout)[(long long)z * cBatch + (long long)m * ldc + n] = v;
            } else if (mode == GOUT_BF16) {
                ((__bf16*)Cout)[(long long)z * cBatch + (long long)m * ldc + n] = (__bf16)v;
            } else { // transposed store (ldc == M)
                ((__bf16*)Cout)[(long long)z * cBatch + (long long)n * ldc + m] = (__bf16)v;
            }
        }
    }
}

// ---------------------------------------------------------------------------
// fp32 -> bf16 weight conversion
// ---------------------------------------------------------------------------
__global__ void k_cvt(const float* __restrict__ src, __bf16* __restrict__ dst, int n) {
    int i = blockIdx.x * blockDim.x + threadIdx.x;
    if (i < n) dst[i] = (__bf16)src[i];
}

// ---------------------------------------------------------------------------
// Gate latents with mask, store fp32 x, LayerNorm -> bf16 q_ln. One block/row.
// ---------------------------------------------------------------------------
__global__ void k_gate_ln(const float* __restrict__ latPrev,   // [256,512]
                          const float* __restrict__ initLat,   // [512]
                          const float* __restrict__ masks,     // [B*T]
                          int t,
                          const float* __restrict__ g, const float* __restrict__ bb,
                          float* __restrict__ Xout,            // [256,512] fp32
                          __bf16* __restrict__ Q) {            // [256,512] bf16
    const int row = blockIdx.x;
    const int tid = threadIdx.x;          // 256 threads
    const float m = masks[row * 64 + t];
    float xv[2], s = 0.0f, s2 = 0.0f;
    #pragma unroll
    for (int i = 0; i < 2; ++i) {
        const int c = tid + i * 256;
        float v = m * latPrev[row * 512 + c] + (1.0f - m) * initLat[c];
        xv[i] = v;
        Xout[row * 512 + c] = v;
        s += v; s2 += v * v;
    }
    __shared__ float r1[256], r2[256];
    r1[tid] = s; r2[tid] = s2;
    __syncthreads();
    for (int o = 128; o; o >>= 1) {
        if (tid < o) { r1[tid] += r1[tid + o]; r2[tid] += r2[tid + o]; }
        __syncthreads();
    }
    const float mean = r1[0] * (1.0f / 512.0f);
    const float var  = r2[0] * (1.0f / 512.0f) - mean * mean;
    const float rstd = rsqrtf(var + 1e-5f);
    #pragma unroll
    for (int i = 0; i < 2; ++i) {
        const int c = tid + i * 256;
        Q[row * 512 + c] = (__bf16)((xv[i] - mean) * rstd * g[c] + bb[c]);
    }
}

// ---------------------------------------------------------------------------
// Generic LayerNorm: fp32 input rows (arbitrary row stride) -> bf16. D in {512,1024}.
// ---------------------------------------------------------------------------
__global__ void k_ln(const float* __restrict__ X, long long xstride,
                     const float* __restrict__ g, const float* __restrict__ bb,
                     __bf16* __restrict__ Y, int D) {
    const int row = blockIdx.x;
    const int tid = threadIdx.x;          // 256 threads
    const float* x = X + (long long)row * xstride;
    const int per = D >> 8;               // 2 or 4
    float xv[4], s = 0.0f, s2 = 0.0f;
    #pragma unroll
    for (int i = 0; i < 4; ++i) {
        if (i < per) {
            float v = x[tid + i * 256];
            xv[i] = v; s += v; s2 += v * v;
        }
    }
    __shared__ float r1[256], r2[256];
    r1[tid] = s; r2[tid] = s2;
    __syncthreads();
    for (int o = 128; o; o >>= 1) {
        if (tid < o) { r1[tid] += r1[tid + o]; r2[tid] += r2[tid + o]; }
        __syncthreads();
    }
    const float invD = 1.0f / (float)D;
    const float mean = r1[0] * invD;
    const float var  = r2[0] * invD - mean * mean;
    const float rstd = rsqrtf(var + 1e-5f);
    #pragma unroll
    for (int i = 0; i < 4; ++i) {
        if (i < per) {
            const int c = tid + i * 256;
            Y[(long long)row * D + c] = (__bf16)((xv[i] - mean) * rstd * g[c] + bb[c]);
        }
    }
}

// ---------------------------------------------------------------------------
// Softmax over 256 columns (one wave per row), fp32 in -> bf16 out.
// ---------------------------------------------------------------------------
__global__ void k_softmax256(const float* __restrict__ S, __bf16* __restrict__ P) {
    const int row  = blockIdx.x * 8 + (threadIdx.x >> 5);
    const int lane = threadIdx.x & 31;
    const float* s = S + (long long)row * 256;
    float e[8], mx = -3.4e38f;
    #pragma unroll
    for (int i = 0; i < 8; ++i) { e[i] = s[lane + i * 32]; mx = fmaxf(mx, e[i]); }
    #pragma unroll
    for (int o = 16; o; o >>= 1) mx = fmaxf(mx, __shfl_xor(mx, o, 32));
    float sum = 0.0f;
    #pragma unroll
    for (int i = 0; i < 8; ++i) { e[i] = expf(e[i] - mx); sum += e[i]; }
    #pragma unroll
    for (int o = 16; o; o >>= 1) sum += __shfl_xor(sum, o, 32);
    const float inv = 1.0f / sum;
    __bf16* p = P + (long long)row * 256;
    #pragma unroll
    for (int i = 0; i < 8; ++i) p[lane + i * 32] = (__bf16)(e[i] * inv);
}

// Duplicate xs -> ls half of d_out (bitwise identical in the reference).
__global__ void k_copy4(const float4* __restrict__ src, float4* __restrict__ dst, int n4) {
    int i = blockIdx.x * blockDim.x + threadIdx.x;
    if (i < n4) dst[i] = src[i];
}

// ---------------------------------------------------------------------------
// Host side
// ---------------------------------------------------------------------------
static void gemm(hipStream_t st,
                 const __bf16* A, int lda, long long aB,
                 const __bf16* W, int ldw, long long wB,
                 const float* bias, const float* resid, int ldr,
                 void* C, int ldc, long long cB,
                 int M, int N, int K, float scale, int mode, int geluF, int zDim) {
    int waves = (M >> 4) * (N >> 6);
    dim3 grid((waves + 7) / 8, 1, zDim);
    k_gemm<<<grid, 256, 0, st>>>(A, lda, aB, W, ldw, wB, bias, resid, ldr,
                                 C, ldc, cB, M, N, K, scale, mode, geluF);
}

extern "C" void kernel_launch(void* const* d_in, const int* in_sizes, int n_in,
                              void* d_out, int out_size, void* d_ws, size_t ws_size,
                              hipStream_t stream) {
    (void)in_sizes; (void)n_in; (void)out_size; (void)ws_size;
    // ---- inputs (setup_inputs dict order, params flattened in insertion order)
    const float* data      = (const float*)d_in[0];   // [16384,1024]
    const float* prevLat   = (const float*)d_in[1];   // [256,8,64] == [256,512]
    const float* masks     = (const float*)d_in[2];   // [16384,1]
    const float* initLat   = (const float*)d_in[3];   // [512]
    const float* ca_lnq_g  = (const float*)d_in[4];
    const float* ca_lnq_b  = (const float*)d_in[5];
    const float* ca_lnkv_g = (const float*)d_in[6];
    const float* ca_lnkv_b = (const float*)d_in[7];
    const float* ca_wq     = (const float*)d_in[8];   // [512,512]
    const float* ca_wk     = (const float*)d_in[9];   // [512,1024]
    const float* ca_wv     = (const float*)d_in[10];  // [512,1024]
    const float* ca_bin    = (const float*)d_in[11];  // [1536]
    const float* ca_wo     = (const float*)d_in[12];
    const float* ca_bo     = (const float*)d_in[13];
    const float* ca_ffg    = (const float*)d_in[14];
    const float* ca_ffb    = (const float*)d_in[15];
    const float* ca_w1     = (const float*)d_in[16];
    const float* ca_b1     = (const float*)d_in[17];
    const float* ca_w2     = (const float*)d_in[18];
    const float* ca_b2     = (const float*)d_in[19];
    const float* sa_ln_g   = (const float*)d_in[20];
    const float* sa_ln_b   = (const float*)d_in[21];
    const float* sa_win    = (const float*)d_in[22];  // [1536,512]
    const float* sa_bin    = (const float*)d_in[23];
    const float* sa_wo     = (const float*)d_in[24];
    const float* sa_bo     = (const float*)d_in[25];
    const float* sa_ffg    = (const float*)d_in[26];
    const float* sa_ffb    = (const float*)d_in[27];
    const float* sa_w1     = (const float*)d_in[28];
    const float* sa_b1     = (const float*)d_in[29];
    const float* sa_w2     = (const float*)d_in[30];
    const float* sa_b2     = (const float*)d_in[31];

    // ---- workspace carve-up
    char* ws = (char*)d_ws;
    size_t off = 0;
    auto alloc = [&](size_t bytes) -> void* {
        void* p = ws + off;
        off += (bytes + 255) & ~(size_t)255;
        return p;
    };
    // bf16 weights
    __bf16* wq_b  = (__bf16*)alloc(262144 * 2);
    __bf16* wk_b  = (__bf16*)alloc(524288 * 2);
    __bf16* wv_b  = (__bf16*)alloc(524288 * 2);
    __bf16* wo_b  = (__bf16*)alloc(262144 * 2);
    __bf16* w1_b  = (__bf16*)alloc(262144 * 2);
    __bf16* w2_b  = (__bf16*)alloc(262144 * 2);
    __bf16* win_b = (__bf16*)alloc(786432 * 2);
    __bf16* swo_b = (__bf16*)alloc(262144 * 2);
    __bf16* sw1_b = (__bf16*)alloc(262144 * 2);
    __bf16* sw2_b = (__bf16*)alloc(262144 * 2);
    // fp32 activations
    float*  xbuf   = (float*)alloc(131072 * 4);   // gated latents (pre-attn residual)
    float*  av     = (float*)alloc(131072 * 4);   // attn + residual (CA)
    float*  x2     = (float*)alloc(131072 * 4);   // CA block output
    float*  av2    = (float*)alloc(131072 * 4);   // attn + residual (SA)
    float*  scores = (float*)alloc(262144 * 4);   // [4,256,256]
    // bf16 activations
    __bf16* q_ln   = (__bf16*)alloc(131072 * 2);
    __bf16* kv_ln  = (__bf16*)alloc(262144 * 2);  // also reused as x_ln (SA)
    __bf16* qp     = (__bf16*)alloc(131072 * 2);
    __bf16* kp     = (__bf16*)alloc(131072 * 2);
    __bf16* vpT    = (__bf16*)alloc(131072 * 2);  // [4,128,256]
    __bf16* attn   = (__bf16*)alloc(262144 * 2);  // [4,256,256]
    __bf16* aout   = (__bf16*)alloc(131072 * 2);  // concat heads [256,512]
    __bf16* hln    = (__bf16*)alloc(131072 * 2);
    __bf16* ff1    = (__bf16*)alloc(131072 * 2);

    // ---- weight conversion (every call: launch must be stateless/deterministic)
    auto cvt = [&](const float* s, __bf16* d, int n) {
        k_cvt<<<(n + 255) / 256, 256, 0, stream>>>(s, d, n);
    };
    cvt(ca_wq, wq_b, 262144);  cvt(ca_wk, wk_b, 524288);  cvt(ca_wv, wv_b, 524288);
    cvt(ca_wo, wo_b, 262144);  cvt(ca_w1, w1_b, 262144);  cvt(ca_w2, w2_b, 262144);
    cvt(sa_win, win_b, 786432);
    cvt(sa_wo, swo_b, 262144); cvt(sa_w1, sw1_b, 262144); cvt(sa_w2, sw2_b, 262144);

    const float scScale = 0.08838834764831843f;  // 1/sqrt(128)
    float* xs = (float*)d_out;                   // [16384,512]
    float* ls = (float*)d_out + (size_t)16384 * 512;

    for (int t = 0; t < 64; ++t) {
        const float* latPrev = (t == 0) ? prevLat : xs + (size_t)(t - 1) * 131072;

        // ===== Cross-attention =====
        k_gate_ln<<<256, 256, 0, stream>>>(latPrev, initLat, masks, t,
                                           ca_lnq_g, ca_lnq_b, xbuf, q_ln);
        // LN over data_t rows: row(batch) = data[batch*64 + t] -> base data + t*1024, stride 64*1024
        k_ln<<<256, 256, 0, stream>>>(data + (size_t)t * 1024, 65536LL,
                                      ca_lnkv_g, ca_lnkv_b, kv_ln, 1024);
        // projections
        gemm(stream, q_ln, 512, 0,  wq_b, 512, 0,  ca_bin + 0,    nullptr, 0,
             qp, 512, 0, 256, 512, 512, 1.0f, GOUT_BF16, 0, 1);
        gemm(stream, kv_ln, 1024, 0, wk_b, 1024, 0, ca_bin + 512,  nullptr, 0,
             kp, 512, 0, 256, 512, 1024, 1.0f, GOUT_BF16, 0, 1);
        gemm(stream, kv_ln, 1024, 0, wv_b, 1024, 0, ca_bin + 1024, nullptr, 0,
             vpT, 256, 0, 256, 512, 1024, 1.0f, GOUT_BF16_TRANS, 0, 1);
        // scores = qp @ kp^T / sqrt(dh), per head (grid.z)
        gemm(stream, qp, 512, 128, kp, 512, 128, nullptr, nullptr, 0,
             scores, 256, 65536, 256, 256, 128, scScale, GOUT_F32, 0, 4);
        k_softmax256<<<128, 256, 0, stream>>>(scores, attn);
        // attn @ V  (V pre-transposed per head: [dh, S])
        gemm(stream, attn, 256, 65536, vpT, 256, 32768, nullptr, nullptr, 0,
             aout, 512, 128, 256, 128, 256, 1.0f, GOUT_BF16, 0, 4);
        // out-proj + residual
        gemm(stream, aout, 512, 0, wo_b, 512, 0, ca_bo, xbuf, 512,
             av, 512, 0, 256, 512, 512, 1.0f, GOUT_F32, 0, 1);
        // FF
        k_ln<<<256, 256, 0, stream>>>(av, 512LL, ca_ffg, ca_ffb, hln, 512);
        gemm(stream, hln, 512, 0, w1_b, 512, 0, ca_b1, nullptr, 0,
             ff1, 512, 0, 256, 512, 512, 1.0f, GOUT_BF16, 1, 1);
        gemm(stream, ff1, 512, 0, w2_b, 512, 0, ca_b2, av, 512,
             x2, 512, 0, 256, 512, 512, 1.0f, GOUT_F32, 0, 1);

        // ===== Self-attention =====
        k_ln<<<256, 256, 0, stream>>>(x2, 512LL, sa_ln_g, sa_ln_b, kv_ln, 512);
        gemm(stream, kv_ln, 512, 0, win_b, 512, 0,              sa_bin + 0,    nullptr, 0,
             qp, 512, 0, 256, 512, 512, 1.0f, GOUT_BF16, 0, 1);
        gemm(stream, kv_ln, 512, 0, win_b + 512 * 512, 512, 0,  sa_bin + 512,  nullptr, 0,
             kp, 512, 0, 256, 512, 512, 1.0f, GOUT_BF16, 0, 1);
        gemm(stream, kv_ln, 512, 0, win_b + 1024 * 512, 512, 0, sa_bin + 1024, nullptr, 0,
             vpT, 256, 0, 256, 512, 512, 1.0f, GOUT_BF16_TRANS, 0, 1);
        gemm(stream, qp, 512, 128, kp, 512, 128, nullptr, nullptr, 0,
             scores, 256, 65536, 256, 256, 128, scScale, GOUT_F32, 0, 4);
        k_softmax256<<<128, 256, 0, stream>>>(scores, attn);
        gemm(stream, attn, 256, 65536, vpT, 256, 32768, nullptr, nullptr, 0,
             aout, 512, 128, 256, 128, 256, 1.0f, GOUT_BF16, 0, 4);
        gemm(stream, aout, 512, 0, swo_b, 512, 0, sa_bo, x2, 512,
             av2, 512, 0, 256, 512, 512, 1.0f, GOUT_F32, 0, 1);
        k_ln<<<256, 256, 0, stream>>>(av2, 512LL, sa_ffg, sa_ffb, hln, 512);
        gemm(stream, hln, 512, 0, sw1_b, 512, 0, sa_b1, nullptr, 0,
             ff1, 512, 0, 256, 512, 512, 1.0f, GOUT_BF16, 1, 1);
        // final: write x directly into xs rows for step t (also next step's latents)
        gemm(stream, ff1, 512, 0, sw2_b, 512, 0, sa_b2, av2, 512,
             xs + (size_t)t * 131072, 512, 0, 256, 512, 512, 1.0f, GOUT_F32, 0, 1);
    }

    // ls is xs reshaped -> duplicate first half into second half
    const int n4 = (16384 * 512) / 4;
    k_copy4<<<(n4 + 255) / 256, 256, 0, stream>>>((const float4*)xs, (float4*)ls, n4);
}
